// GameURMAttention_42855183680160
// MI455X (gfx1250) — matrix operand order repaired and checked by reading, not compiled, not run on hardware
//
#include <hip/hip_runtime.h>

#define HIDDEN   512
#define HEADS    8
#define HEAD_DIM 64
#define SEQ      16
#define BATCH    4096
#define NQKV     (3 * HIDDEN)

typedef __attribute__((ext_vector_type(16))) _Float16 v16h;
typedef __attribute__((ext_vector_type(8)))  _Float16 v8h;
typedef __attribute__((ext_vector_type(8)))  float    v8f;

typedef __attribute__((address_space(3))) _Float16 lds_f16;

// ---------------------------------------------------------------------------
// WMMA fragment loaders (wave32; k = 16*(lane>>4) + s, not the CDNA5 ISA §7.12.2 table order)
// ---------------------------------------------------------------------------

// A-matrix 16x32 f16 (MxK), row-major source with leading dim `ld`.
__device__ __forceinline__ v16h load_a_frag(const _Float16* tile, int ld, int lane) {
  const int m  = lane & 15;
  const int hf = lane >> 4;
  const _Float16* p = tile + m * ld + 16 * hf;
  v8h lo = *(const v8h*)(p);        // K = 16*hf .. +7
  v8h hi = *(const v8h*)(p + 8);    // K = 16*hf+8 .. +15
  return __builtin_shufflevector(lo, hi, 0,1,2,3,4,5,6,7,8,9,10,11,12,13,14,15);
}

// B-matrix 32x16 f16 (KxN) with B[k][n] = M[n][k] for row-major M (stride ld).
__device__ __forceinline__ v16h load_bT_frag(const _Float16* tile, int ld, int lane) {
  const int n  = lane & 15;
  const int hf = lane >> 4;
  const _Float16* p = tile + n * ld + 16 * hf;
  v8h lo = *(const v8h*)(p);        // K = 16*hf .. +7
  v8h hi = *(const v8h*)(p + 8);    // K = 16*hf+8 .. +15
  return __builtin_shufflevector(lo, hi, 0,1,2,3,4,5,6,7,8,9,10,11,12,13,14,15);
}

__device__ __forceinline__ v8f wmma_f16(v16h a, v16h b, v8f c) {
  return __builtin_amdgcn_wmma_f32_16x16x32_f16(
      false, a, false, b, (short)0, c, false, false);
}

// ---------------------------------------------------------------------------
// K0: convert + swizzle weights f32 -> f16 in B-fragment-major order.
// Fragment id f = tile*16 + kc; per lane (32B): W[t*16 + (lane&15)][kc*32 +
// 16*(lane>>4) + 0..15]. One thread per (f,lane), 16 elements each.
// ---------------------------------------------------------------------------
__global__ __launch_bounds__(256) void cvt_swizzle_kernel(
    const float* __restrict__ wqkv, const float* __restrict__ wo,
    _Float16* __restrict__ wqkvsw, _Float16* __restrict__ wosw) {
  int u = blockIdx.x * 256 + threadIdx.x;           // 0 .. 65535
  const float* src;
  _Float16* dst;
  int lf;
  if (u < 49152) { lf = u;         src = wqkv; dst = wqkvsw; }
  else           { lf = u - 49152; src = wo;   dst = wosw;   }
  int lane = lf & 31;
  int f    = lf >> 5;
  int kc   = f & 15;
  int t    = f >> 4;
  int n    = t * 16 + (lane & 15);
  int k0   = kc * 32 + 16 * (lane >> 4);
  const float* s = src + (size_t)n * HIDDEN + k0;
  _Float16* d = dst + (size_t)lf * 16;
#pragma unroll
  for (int e = 0; e < 16; ++e) d[e] = (_Float16)s[e];
}

// fragment load from a swizzled weight array (one contiguous 32B per lane)
__device__ __forceinline__ v16h load_w_frag(const v16h* wsw, int tile, int kc, int lane) {
  return wsw[(size_t)(tile * 16 + kc) * 32 + lane];
}

// ---------------------------------------------------------------------------
// K1: fused QKV projection + attention, one block (8 waves) per batch element.
// LDS: Xb 16KB | Uni 32KB (Q,K then reused as V^T) | Pb 8KB  = 56KB
// ---------------------------------------------------------------------------
__global__ __launch_bounds__(256) void qkv_attn_kernel(
    const float* __restrict__ x, const _Float16* __restrict__ wqkvsw,
    _Float16* __restrict__ attnh) {
  __shared__ _Float16 Xb[SEQ * HIDDEN];   // x tile, f16, 16x512
  __shared__ _Float16 Uni[8 * 64 * 32];   // Q[8][16][64] | K[8][16][64]  -> later Vt[8][64][32]
  __shared__ _Float16 Pb[8 * 16 * 32];    // scores -> probs, padded to K=32

  const int tid  = threadIdx.x;
  const int lane = tid & 31;
  const int w    = tid >> 5;          // wave id 0..7
  const int hf   = lane >> 4;
  const int nl   = lane & 15;
  const int b    = blockIdx.x;
  const v16h* wq = (const v16h*)wqkvsw;

  // ---- stage X (f32 -> f16) into LDS ----
  {
    const float4* xs = (const float4*)(x + (size_t)b * SEQ * HIDDEN);
#pragma unroll
    for (int it = 0; it < 8; ++it) {
      int idx  = it * 256 + tid;
      float4 v = xs[idx];
      _Float16* d = Xb + idx * 4;
      d[0] = (_Float16)v.x; d[1] = (_Float16)v.y;
      d[2] = (_Float16)v.z; d[3] = (_Float16)v.w;
    }
  }
  __syncthreads();

  // ---- phase 1: Q and K projections (tiles 0..63, 8 per wave),
  //      software-pipelined in two fragment groups of 4 ----
  {
    v8f acc[8] = {};
    v16h cur[4];
#pragma unroll
    for (int i = 0; i < 4; ++i) cur[i] = load_w_frag(wq, w * 8 + i, 0, lane);
    for (int kc = 0; kc < 16; ++kc) {
      v16h a = load_a_frag(Xb + 32 * kc, HIDDEN, lane);
      v16h oth[4];
#pragma unroll
      for (int i = 0; i < 4; ++i) oth[i] = load_w_frag(wq, w * 8 + 4 + i, kc, lane);
#pragma unroll
      for (int i = 0; i < 4; ++i) acc[i] = wmma_f16(a, cur[i], acc[i]);
      v16h nxt[4];
      const int kn = (kc + 1) & 15;   // wrap: redundant reload at kc=15, harmless
#pragma unroll
      for (int i = 0; i < 4; ++i) nxt[i] = load_w_frag(wq, w * 8 + i, kn, lane);
#pragma unroll
      for (int i = 0; i < 4; ++i) acc[4 + i] = wmma_f16(a, oth[i], acc[4 + i]);
#pragma unroll
      for (int i = 0; i < 4; ++i) cur[i] = nxt[i];
    }
#pragma unroll
    for (int i = 0; i < 8; ++i) {
      int t    = w * 8 + i;
      int c    = (t * 16) & 511;
      int head = c >> 6;
      int d0   = c & 63;
      _Float16* dst = Uni + (t >= 32 ? 8192 : 0) + head * 1024 + d0;
#pragma unroll
      for (int r = 0; r < 8; ++r) {
        int m = r + 8 * hf;
        dst[m * 64 + nl] = (_Float16)acc[i][r];
      }
    }
  }
  __syncthreads();

  // ---- phase 2: scores = Q @ K^T for head w -> Pb (f16) ----
  {
    v8f s = {};
#pragma unroll
    for (int kc = 0; kc < 2; ++kc) {
      v16h qa = load_a_frag(Uni + w * 1024 + 32 * kc, 64, lane);
      v16h kb = load_bT_frag(Uni + 8192 + w * 1024 + 32 * kc, 64, lane);
      s = wmma_f16(qa, kb, s);
    }
#pragma unroll
    for (int r = 0; r < 8; ++r) {
      int m = r + 8 * hf;
      Pb[(w * 16 + m) * 32 + nl] = (_Float16)s[r];
    }
  }
  __syncthreads();

  // ---- phase 3: softmax (scale 1/sqrt(64)) ----
  if (tid < 128) {
    int hh = tid >> 4, i = tid & 15;
    _Float16* prow = Pb + (hh * 16 + i) * 32;
    float rowv[16];
    float mx = -1e30f;
#pragma unroll
    for (int j = 0; j < 16; ++j) {
      rowv[j] = 0.125f * (float)prow[j];
      mx = fmaxf(mx, rowv[j]);
    }
    float sum = 0.0f;
#pragma unroll
    for (int j = 0; j < 16; ++j) { rowv[j] = __expf(rowv[j] - mx); sum += rowv[j]; }
    float inv = 1.0f / sum;
#pragma unroll
    for (int j = 0; j < 16; ++j) prow[j] = (_Float16)(rowv[j] * inv);
#pragma unroll
    for (int j = 16; j < 32; ++j) prow[j] = (_Float16)0.0f;
  }
  __syncthreads();

  // ---- phase 4: V projection -> Vt[8][64][32] (cols 16..31 zeroed),
  //      double-buffered over kc ----
  {
    for (int idx = tid; idx < 512; idx += 256) {   // idx = head*64 + d
      v8h z = {};
      *(v8h*)(Uni + idx * 32 + 16) = z;
      *(v8h*)(Uni + idx * 32 + 24) = z;
    }
    v8f acc[4] = {};
    v16h cur[4];
#pragma unroll
    for (int i = 0; i < 4; ++i) cur[i] = load_w_frag(wq, 64 + w * 4 + i, 0, lane);
    for (int kc = 0; kc < 16; ++kc) {
      v16h a = load_a_frag(Xb + 32 * kc, HIDDEN, lane);
      v16h nxt[4];
      const int kn = (kc + 1) & 15;
#pragma unroll
      for (int i = 0; i < 4; ++i) nxt[i] = load_w_frag(wq, 64 + w * 4 + i, kn, lane);
#pragma unroll
      for (int i = 0; i < 4; ++i) acc[i] = wmma_f16(a, cur[i], acc[i]);
#pragma unroll
      for (int i = 0; i < 4; ++i) cur[i] = nxt[i];
    }
#pragma unroll
    for (int i = 0; i < 4; ++i) {
      int vcol0 = (64 + w * 4 + i) * 16 - 1024;    // 0..511
      int head  = vcol0 >> 6;
      int d0    = vcol0 & 63;
#pragma unroll
      for (int r = 0; r < 8; ++r) {
        int m = r + 8 * hf;                        // seq index j
        Uni[head * 2048 + (d0 + nl) * 32 + m] = (_Float16)acc[i][r];
      }
    }
  }
  __syncthreads();

  // ---- phase 5: attn = P @ V for head w ----
  {
    v16h pa = load_a_frag(Pb + w * 512, 32, lane);
    _Float16* arow = attnh + (size_t)b * SEQ * HIDDEN + w * 64;
#pragma unroll
    for (int i = 0; i < 4; ++i) {
      v8f acc = {};
      v16h vb = load_bT_frag(Uni + w * 2048 + i * 512, 32, lane);
      acc = wmma_f16(pa, vb, acc);
#pragma unroll
      for (int r = 0; r < 8; ++r) {
        int m = r + 8 * hf;
        arow[(size_t)m * HIDDEN + i * 16 + nl] = (_Float16)acc[r];
      }
    }
  }
}

// ---------------------------------------------------------------------------
// K2: output projection  out = attn_h @ Wo^T (f32 out).
// A strip staged to LDS via async global->LDS DMA; B double-buffered.
// ---------------------------------------------------------------------------
__global__ __launch_bounds__(256) void outproj_kernel(
    const _Float16* __restrict__ attnh, const _Float16* __restrict__ wosw,
    float* __restrict__ out) {
  __shared__ _Float16 Ab[SEQ * HIDDEN];   // 16KB

  const int tid  = threadIdx.x;
  const int lane = tid & 31;
  const int w    = tid >> 5;
  const int hf   = lane >> 4;
  const int nl   = lane & 15;
  const size_t row0 = (size_t)blockIdx.x * SEQ;
  const v16h* wo = (const v16h*)wosw;

  // ---- async stage: 16KB in 1024 x 16B lane transfers (4 per thread) ----
  {
    unsigned lbase = (unsigned)(__UINTPTR_TYPE__)(lds_f16*)Ab;   // LDS byte offset
    const _Float16* g0 = attnh + row0 * HIDDEN;
#pragma unroll
    for (int it = 0; it < 4; ++it) {
      int idx = it * 256 + tid;                    // 16B chunk id
      unsigned laddr = lbase + (unsigned)idx * 16u;
      const _Float16* gaddr = g0 + (size_t)idx * 8;
      asm volatile("global_load_async_to_lds_b128 %0, %1, off"
                   :: "v"(laddr), "v"(gaddr) : "memory");
    }
    asm volatile("s_wait_asynccnt 0" ::: "memory");
  }
  __syncthreads();

  v8f acc[4] = {};
  v16h cur[4];
#pragma unroll
  for (int i = 0; i < 4; ++i) cur[i] = load_w_frag(wo, w * 4 + i, 0, lane);
  for (int kc = 0; kc < 16; ++kc) {
    v16h a = load_a_frag(Ab + 32 * kc, HIDDEN, lane);
    v16h nxt[4];
    const int kn = (kc + 1) & 15;
#pragma unroll
    for (int i = 0; i < 4; ++i) nxt[i] = load_w_frag(wo, w * 4 + i, kn, lane);
#pragma unroll
    for (int i = 0; i < 4; ++i) acc[i] = wmma_f16(a, cur[i], acc[i]);
#pragma unroll
    for (int i = 0; i < 4; ++i) cur[i] = nxt[i];
  }
#pragma unroll
  for (int i = 0; i < 4; ++i) {
#pragma unroll
    for (int r = 0; r < 8; ++r) {
      int m = r + 8 * hf;
      out[(row0 + m) * HIDDEN + w * 64 + i * 16 + nl] = acc[i][r];
    }
  }
}

// ---------------------------------------------------------------------------
// launch
// ---------------------------------------------------------------------------
extern "C" void kernel_launch(void* const* d_in, const int* in_sizes, int n_in,
                              void* d_out, int out_size, void* d_ws, size_t ws_size,
                              hipStream_t stream) {
  const float* x    = (const float*)d_in[0];   // [4096,16,512]
  const float* wqkv = (const float*)d_in[1];   // [1536,512]
  const float* wo   = (const float*)d_in[2];   // [512,512]
  float* out        = (float*)d_out;           // [4096,16,512]

  // ws layout (f16): wqkvsw [0,1.5MB) | wosw [1.5MB,2MB) | attn [2MB, 2MB+64MB)
  _Float16* wqkvsw = (_Float16*)d_ws;
  _Float16* wosw   = (_Float16*)((char*)d_ws + (size_t)NQKV * HIDDEN * 2);
  _Float16* attnh  = (_Float16*)((char*)d_ws + ((size_t)2 << 20));

  cvt_swizzle_kernel<<<256, 256, 0, stream>>>(wqkv, wo, wqkvsw, wosw);
  qkv_attn_kernel<<<BATCH, 256, 0, stream>>>(x, wqkvsw, attnh);
  outproj_kernel<<<BATCH, 256, 0, stream>>>(attnh, wosw, out);
}
